// kv_cache_multihead_78812649882269
// MI455X (gfx1250) — matrix operand order repaired and checked by reading, not compiled, not run on hardware
//
#include <hip/hip_runtime.h>

// ---------------------------------------------------------------------------
// Problem constants (from reference): E=1024, H=16, D=64, B=8, T=64,
// cache capacity 8192, start_pos=4096 (read from d_in[7] at runtime).
// ---------------------------------------------------------------------------
constexpr int kE   = 1024;
constexpr int kH   = 16;
constexpr int kD   = 64;
constexpr int kB   = 8;
constexpr int kT   = 64;
constexpr int kCap = 8192;
constexpr int kParts = 4;     // S-dimension split of the attention grid

typedef _Float16 half_t;
typedef __attribute__((ext_vector_type(16))) _Float16 v16h;
typedef __attribute__((ext_vector_type(8)))  float    v8f;
typedef __attribute__((ext_vector_type(4)))  unsigned int u32x4;
typedef __attribute__((ext_vector_type(8)))  int          i32x8;
typedef __attribute__((ext_vector_type(4)))  int          i32x4;

struct __align__(16) h8 { _Float16 v[8]; };

// D = A(16x32 f16) * B(32x16 f16) + C(16x16 f32)
__device__ __forceinline__ v8f wmma_f16(v16h a, v16h b, v8f c) {
  return __builtin_amdgcn_wmma_f32_16x16x32_f16(
      /*neg_a=*/false, a, /*neg_b=*/false, b,
      /*c_mod=*/(short)0, c, /*reuse_a=*/false, /*reuse_b=*/false);
}

// A fragment (this code's contiguous k order): lane<16 holds row (lane&15),
// K = base..base+15; lane>=16 same row, K=16..31.
// Caller passes p = rowptr + k0 + (lane>>4)*16.
__device__ __forceinline__ v16h load_a_frag(const half_t* p) {
  h8 a0 = *(const h8*)(p);
  h8 a1 = *(const h8*)(p + 8);
  v16h a;
#pragma unroll
  for (int i = 0; i < 8; ++i) { a[i] = a0.v[i]; a[i + 8] = a1.v[i]; }
  return a;
}

// B fragment (this code's contiguous k order): lane<16 = column (lane&15), K=0..15; lane>=16 same
// column, K=16..31. Caller passes p = colptr + k0 + (lane>>4)*16.
__device__ __forceinline__ v16h load_b_frag(const half_t* p) {
  h8 b0 = *(const h8*)(p);
  h8 b1 = *(const h8*)(p + 8);
  v16h b;
#pragma unroll
  for (int i = 0; i < 8; ++i) { b[i] = b0.v[i]; b[i + 8] = b1.v[i]; }
  return b;
}

__device__ __forceinline__ v16h load_b_frag_f32(const float* p) {
  v16h b;
#pragma unroll
  for (int i = 0; i < 16; ++i) b[i] = (half_t)p[i];
  return b;
}

// Reductions across the 16 lanes of a C-fragment row half (xor masks < 16).
__device__ __forceinline__ float hmax16(float v) {
#pragma unroll
  for (int m = 1; m <= 8; m <<= 1) v = fmaxf(v, __shfl_xor(v, m, 32));
  return v;
}
__device__ __forceinline__ float hsum16(float v) {
#pragma unroll
  for (int m = 1; m <= 8; m <<= 1) v += __shfl_xor(v, m, 32);
  return v;
}

// ---------------------------------------------------------------------------
// TDM: DMA a 2D f32 tile (64 contiguous elements x `rows` rows, row stride
// `stride_elems` elements) from global memory into LDS at byte offset
// `lds_off`. D# bit layout per CDNA5 ISA 8.3/8.4 (count=1, type=2 "image",
// data_size=4B, no padding/iterate/gather/multicast). Tracked by TENSORcnt;
// per-wave TDM ops complete in order, enabling threshold waits.
// ---------------------------------------------------------------------------
__device__ __forceinline__ void tdm_load_2d_f32(unsigned int lds_off,
                                                const float* gptr,
                                                int rows, int stride_elems) {
  unsigned long long ga = (unsigned long long)(uintptr_t)gptr;
  u32x4 g0;
  g0[0] = 1u;                                            // count=1, user mode
  g0[1] = lds_off;                                       // lds_addr (bytes)
  g0[2] = (unsigned int)ga;                              // global_addr[31:0]
  g0[3] = (unsigned int)((ga >> 32) & 0x1FFFFFFull)      // global_addr[56:32]
          | 0x80000000u;                                 // type=2 (bits 127:126)
  i32x8 g1;
  g1[0] = (int)(2u << 16);                 // data_size=2 (4 bytes), mask=0
  g1[1] = (int)(64u << 16);                // tensor_dim0 = 64 (low 16 bits)
  g1[2] = (int)((unsigned)rows << 16);     // tensor_dim1[15:0]
  g1[3] = (int)(64u << 16);                // tile_dim0 = 64
  g1[4] = rows;                            // tile_dim1
  g1[5] = stride_elems;                    // tensor_dim0_stride[31:0]
  g1[6] = 0;
  g1[7] = 0;
  i32x4 z4 = {0, 0, 0, 0};                 // groups 2/3 unused (2D tile)
  i32x8 z8 = {0, 0, 0, 0, 0, 0, 0, 0};
  __builtin_amdgcn_tensor_load_to_lds(g0, g1, z4, z4, z8, 0);
}

// ---------------------------------------------------------------------------
// Precision/layout prep: x -> f16; W -> f16 transposed (N-major).
// ---------------------------------------------------------------------------
__global__ void cvt_x_kernel(const float* __restrict__ x, half_t* __restrict__ xb) {
  int i = blockIdx.x * blockDim.x + threadIdx.x;
  if (i < kB * kT * kE) xb[i] = (half_t)x[i];
}

__global__ void transpose_w_kernel(const float* __restrict__ W, half_t* __restrict__ WT) {
  int i = blockIdx.x * blockDim.x + threadIdx.x;
  if (i < kE * kE) {
    int n = i >> 10, k = i & (kE - 1);
    WT[i] = (half_t)W[k * kE + n];  // WT[n][k] = W[k][n]
  }
}

// ---------------------------------------------------------------------------
// Fused QKV projection + buggy RoPE (angle indexed by HEAD, broadcast over
// time, exactly as the reference) + repack to (B,H,T,D) f16.
// ---------------------------------------------------------------------------
__global__ void __launch_bounds__(128)
qkv_rope_kernel(const half_t* __restrict__ xb,
                const half_t* __restrict__ wqT,
                const half_t* __restrict__ wkT,
                const half_t* __restrict__ wvT,
                half_t* __restrict__ q_r,
                half_t* __restrict__ k_r,
                half_t* __restrict__ v_r,
                const int* __restrict__ spp) {
  const int w = threadIdx.x >> 5, lane = threadIdx.x & 31;
  const int lrow = lane & 15, khalf = lane >> 4;
  const int m0 = blockIdx.x * 16;
  const int h  = blockIdx.y;
  const int mat = blockIdx.z;
  const half_t* WT = (mat == 0) ? wqT : (mat == 1) ? wkT : wvT;
  const int n0 = h * kD + w * 16;

  const half_t* arow = xb + (size_t)(m0 + lrow) * kE;
  const half_t* brow = WT + (size_t)(n0 + lrow) * kE;

  v8f c = {};
#pragma unroll 4
  for (int k0 = 0; k0 < kE; k0 += 32) {
    v16h a = load_a_frag(arow + k0 + khalf * 16);
    v16h b = load_b_frag(brow + k0 + khalf * 16);
    c = wmma_f16(a, b, c);
  }

  const int sp = *spp;
  const bool dorope = (mat < 2);
  const int dcol = w * 16 + lrow;           // 0..63 inside this head
  float cs = 1.f, sn = 0.f;
  if (dorope) {
    const int f = dcol >> 1;
    // inv_freq = 10000^(-f/32) = exp(-f * ln(1e4)/32)
    const float inv = __expf(-0.28782313662425573f * (float)f);
    const float ang = (float)(sp + h) * inv;   // head-indexed position (bug)
    cs = __cosf(ang);
    sn = __sinf(ang);
  }
  half_t* dst = (mat == 0) ? q_r : (mat == 1) ? k_r : v_r;
#pragma unroll
  for (int r = 0; r < 8; ++r) {
    float val = c[r];
    float other = __shfl_xor(val, 1, 32);    // pair partner (col ^ 1)
    float o = dorope ? ((lane & 1) ? (other * sn + val * cs)
                                   : (val * cs - other * sn))
                     : val;
    const int m = m0 + khalf * 8 + r;        // global row = b*T + t
    const int bb = m >> 6, t = m & 63;
    dst[(((size_t)(bb * kH + h) * kT + t) * kD) + dcol] = (half_t)o;
  }
}

// ---------------------------------------------------------------------------
// Flash-attention partial: grid (H, B, 4 S-parts) = 512 blocks of 128 threads
// (4 waves = 4 query subtiles). Each 32-key chunk of K/V is staged once into
// LDS with double-buffered TDM (manual f16->f32 staging for the appended
// tokens), overlapping the next chunk's DMA with the current chunk's WMMAs.
// Emits un-normalized partials (m, l, acc) per part to workspace.
// ---------------------------------------------------------------------------
__global__ void __launch_bounds__(128)
attn_part_kernel(const half_t* __restrict__ q_r,
                 const half_t* __restrict__ k_r,
                 const half_t* __restrict__ v_r,
                 const float* __restrict__ cache_k,
                 const float* __restrict__ cache_v,
                 float* __restrict__ pacc,
                 float* __restrict__ pml,
                 const int* __restrict__ spp) {
  __shared__ __align__(16) float lds_k[2][32 * 64];     // 16 KB (x2 buffers)
  __shared__ __align__(16) float lds_v[2][32 * 64];     // 16 KB (x2 buffers)
  __shared__ __align__(16) half_t lds_p[4][16][32];     // 4 KB: P staging

  const int h = blockIdx.x, b = blockIdx.y, part = blockIdx.z;
  const int w = threadIdx.x >> 5, lane = threadIdx.x & 31;
  const int lrow = lane & 15, khalf = lane >> 4;
  const int mi = w;                          // query subtile

  const int sp = *spp;                       // 4096
  const int nch_total = (sp + kT) / 32;      // 130 chunks of 32 keys
  const int base_c = nch_total >> 2;         // 32
  const int rem    = nch_total & 3;          // 2
  const int c0  = part * base_c + (part < rem ? part : rem);
  const int cnt = base_c + (part < rem ? 1 : 0);

  // Stage chunk `ci` into buffer `nbuf`: TDM for the f32 history cache,
  // manual f16->f32 conversion for the appended tokens.
  auto stage = [&](int nbuf, int ci) {
    const int s0 = ci * 32;
    if (s0 < sp) {
      if (w == 0) {
        const size_t base = (((size_t)(b * kCap + s0)) * kH + h) * kD;
        tdm_load_2d_f32((unsigned int)(uintptr_t)&lds_k[nbuf][0],
                        cache_k + base, 32, kH * kD);
        tdm_load_2d_f32((unsigned int)(uintptr_t)&lds_v[nbuf][0],
                        cache_v + base, 32, kH * kD);
      }
    } else {
      const int tid = threadIdx.x;           // 0..127
      const int key = tid >> 2, dp = (tid & 3) * 16;
      const size_t src = (size_t)((b * kH + h) * kT + (s0 - sp + key)) * kD + dp;
      const half_t* ks = k_r + src;
      const half_t* vs = v_r + src;
      float* dk = &lds_k[nbuf][key * 64 + dp];
      float* dv = &lds_v[nbuf][key * 64 + dp];
#pragma unroll
      for (int i = 0; i < 16; ++i) { dk[i] = (float)ks[i]; dv[i] = (float)vs[i]; }
    }
  };

  const half_t* qbase = q_r + (size_t)((b * kH + h) * kT) * kD;
  const v16h aq0 = load_a_frag(qbase + (mi * 16 + lrow) * kD + 0  + khalf * 16);
  const v16h aq1 = load_a_frag(qbase + (mi * 16 + lrow) * kD + 32 + khalf * 16);

  float m_[8], l_[8];
  v8f acc[4];
  v8f zero = {};
#pragma unroll
  for (int r = 0; r < 8; ++r) { m_[r] = -1.0e30f; l_[r] = 0.f; }
#pragma unroll
  for (int dd = 0; dd < 4; ++dd) acc[dd] = zero;

  stage(0, c0);                              // prologue: fill buffer 0

  for (int i = 0; i < cnt; ++i) {
    const int buf = i & 1;
    const int s0 = (c0 + i) * 32;
    const bool next_tdm = (i + 1 < cnt) && ((c0 + i + 1) * 32 < sp);

    if (i + 1 < cnt) stage(buf ^ 1, c0 + i + 1);   // prefetch next chunk

    // Per-wave TDM ops complete in order: the 2 newest pending ops belong to
    // chunk i+1, so tensorcnt<=2 means chunk i's tiles have landed.
    if (next_tdm) __builtin_amdgcn_s_wait_tensorcnt(2);
    else          __builtin_amdgcn_s_wait_tensorcnt(0);
    __syncthreads();                         // chunk i visible to all waves

    // ---- scores: Q(16x64) . K_chunk^T(64x32) -> two 16x16 C-frags ----
    v8f sf0 = zero, sf1 = zero;
#pragma unroll
    for (int j = 0; j < 2; ++j) {
      v8f sc = zero;
#pragma unroll
      for (int ds = 0; ds < 2; ++ds) {
        const v16h bk =
            load_b_frag_f32(&lds_k[buf][(j * 16 + lrow) * 64 + ds * 32 + khalf * 16]);
        sc = wmma_f16(ds ? aq1 : aq0, bk, sc);
      }
      if (j == 0) sf0 = sc; else sf1 = sc;
    }

    // ---- scale + causal mask (only appended positions are ever masked) ----
#pragma unroll
    for (int r = 0; r < 8; ++r) {
      const int trow = mi * 16 + khalf * 8 + r;
      float s0v = sf0[r] * 0.125f;
      float s1v = sf1[r] * 0.125f;
      const int sg0 = s0 + lrow;
      const int sg1 = s0 + 16 + lrow;
      if (sg0 >= sp && (sg0 - sp) > trow) s0v = -3.0e38f;
      if (sg1 >= sp && (sg1 - sp) > trow) s1v = -3.0e38f;
      sf0[r] = s0v;
      sf1[r] = s1v;
    }

    // ---- online softmax (row stats replicated across each 16-lane half) ----
#pragma unroll
    for (int r = 0; r < 8; ++r) {
      const float cm = hmax16(fmaxf(sf0[r], sf1[r]));
      const float nm = fmaxf(m_[r], cm);
      const float corr = __expf(m_[r] - nm);
      const float p0 = __expf(sf0[r] - nm);
      const float p1 = __expf(sf1[r] - nm);
      l_[r] = l_[r] * corr + hsum16(p0 + p1);
      m_[r] = nm;
      sf0[r] = p0;
      sf1[r] = p1;
#pragma unroll
      for (int dd = 0; dd < 4; ++dd) acc[dd][r] *= corr;
    }

    // ---- stage P (C-layout) through LDS, reload as A-fragment ----
#pragma unroll
    for (int r = 0; r < 8; ++r) {
      lds_p[mi][khalf * 8 + r][lrow]      = (half_t)sf0[r];
      lds_p[mi][khalf * 8 + r][16 + lrow] = (half_t)sf1[r];
    }
    const v16h ap = load_a_frag(&lds_p[mi][lrow][khalf * 16]);  // in-order DS

    // ---- ctx += P(16x32) . V_chunk(32x64) ----
#pragma unroll
    for (int dd = 0; dd < 4; ++dd) {
      const int dcol = dd * 16 + lrow;
      v16h bv;
#pragma unroll
      for (int i2 = 0; i2 < 16; ++i2)
        bv[i2] = (half_t)lds_v[buf][(khalf * 16 + i2) * 64 + dcol];
      acc[dd] = wmma_f16(ap, bv, acc[dd]);
    }

    __syncthreads();   // all waves done reading buf before it is re-staged
  }

  // ---- emit un-normalized partials for this S-part ----
  const int bh = b * kH + h;
#pragma unroll
  for (int r = 0; r < 8; ++r) {
    const int trow = mi * 16 + khalf * 8 + r;
    const size_t rowb = (size_t)(bh * kParts + part) * kT + trow;
    if (lrow == 0) { pml[rowb * 2 + 0] = m_[r]; pml[rowb * 2 + 1] = l_[r]; }
#pragma unroll
    for (int dd = 0; dd < 4; ++dd)
      pacc[rowb * 64 + dd * 16 + lrow] = acc[dd][r];
  }
}

// ---------------------------------------------------------------------------
// Merge the 4 S-part partials (log-sum-exp combine), normalize, emit ctx f16.
// One block per (b,h); thread = (t, 16-wide d group).
// ---------------------------------------------------------------------------
__global__ void __launch_bounds__(256)
attn_merge_kernel(const float* __restrict__ pacc,
                  const float* __restrict__ pml,
                  half_t* __restrict__ ctx) {
  const int h = blockIdx.x, b = blockIdx.y;
  const int bh = b * kH + h;
  const int t  = threadIdx.x >> 2;
  const int dq = threadIdx.x & 3;

  float mp[kParts], lp[kParts];
  float mm = -3.4e38f;
#pragma unroll
  for (int p = 0; p < kParts; ++p) {
    const size_t rowb = (size_t)(bh * kParts + p) * kT + t;
    mp[p] = pml[rowb * 2 + 0];
    lp[p] = pml[rowb * 2 + 1];
    mm = fmaxf(mm, mp[p]);
  }
  float L = 0.f;
  float o[16];
#pragma unroll
  for (int d = 0; d < 16; ++d) o[d] = 0.f;
#pragma unroll
  for (int p = 0; p < kParts; ++p) {
    const float wgt = __expf(mp[p] - mm);
    L += lp[p] * wgt;
    const float* pa = &pacc[((size_t)(bh * kParts + p) * kT + t) * 64 + dq * 16];
#pragma unroll
    for (int d = 0; d < 16; ++d) o[d] += pa[d] * wgt;
  }
  const float invL = 1.0f / L;
  half_t* dst = &ctx[(size_t)(b * kT + t) * kE + h * kD + dq * 16];
#pragma unroll
  for (int d = 0; d < 16; ++d) dst[d] = (half_t)(o[d] * invL);
}

// ---------------------------------------------------------------------------
// Output projection: ctx(512x1024 f16) @ Wo -> out f32 (B,T,E).
// ---------------------------------------------------------------------------
__global__ void __launch_bounds__(128)
out_gemm_kernel(const half_t* __restrict__ ctx,
                const half_t* __restrict__ woT,
                float* __restrict__ out) {
  const int w = threadIdx.x >> 5, lane = threadIdx.x & 31;
  const int lrow = lane & 15, khalf = lane >> 4;
  const int m0 = blockIdx.x * 16;
  const int n0 = blockIdx.y * 64 + w * 16;
  const half_t* arow = ctx + (size_t)(m0 + lrow) * kE;
  const half_t* brow = woT + (size_t)(n0 + lrow) * kE;
  v8f c = {};
#pragma unroll 4
  for (int k0 = 0; k0 < kE; k0 += 32) {
    v16h a = load_a_frag(arow + k0 + khalf * 16);
    v16h b = load_b_frag(brow + k0 + khalf * 16);
    c = wmma_f16(a, b, c);
  }
#pragma unroll
  for (int r = 0; r < 8; ++r) {
    const int m = m0 + khalf * 8 + r;
    out[(size_t)m * kE + n0 + lrow] = c[r];
  }
}

// ---------------------------------------------------------------------------
extern "C" void kernel_launch(void* const* d_in, const int* in_sizes, int n_in,
                              void* d_out, int out_size, void* d_ws, size_t ws_size,
                              hipStream_t stream) {
  (void)in_sizes; (void)n_in; (void)out_size; (void)ws_size;
  const float* x  = (const float*)d_in[0];
  const float* ck = (const float*)d_in[1];
  const float* cv = (const float*)d_in[2];
  const float* Wq = (const float*)d_in[3];
  const float* Wk = (const float*)d_in[4];
  const float* Wv = (const float*)d_in[5];
  const float* Wo = (const float*)d_in[6];
  const int*   sp = (const int*)d_in[7];
  float* out = (float*)d_out;

  // Workspace: ~13 MB f16 tensors + ~8.3 MB f32 attention partials.
  half_t* ws  = (half_t*)d_ws;
  half_t* xb  = ws;                                  // 512*1024
  half_t* wqT = xb  + (size_t)kB * kT * kE;          // 1024*1024 each
  half_t* wkT = wqT + (size_t)kE * kE;
  half_t* wvT = wkT + (size_t)kE * kE;
  half_t* woT = wvT + (size_t)kE * kE;
  half_t* q_r = woT + (size_t)kE * kE;               // (B,H,T,D) each
  half_t* k_r = q_r + (size_t)kB * kH * kT * kD;
  half_t* v_r = k_r + (size_t)kB * kH * kT * kD;
  half_t* ctx = v_r + (size_t)kB * kH * kT * kD;     // (B*T, E)
  float*  pacc = (float*)(ctx + (size_t)kB * kT * kE);       // BH*4*T*D
  float*  pml  = pacc + (size_t)kB * kH * kParts * kT * kD;  // BH*4*T*2

  cvt_x_kernel<<<(kB * kT * kE + 255) / 256, 256, 0, stream>>>(x, xb);
  transpose_w_kernel<<<(kE * kE + 255) / 256, 256, 0, stream>>>(Wq, wqT);
  transpose_w_kernel<<<(kE * kE + 255) / 256, 256, 0, stream>>>(Wk, wkT);
  transpose_w_kernel<<<(kE * kE + 255) / 256, 256, 0, stream>>>(Wv, wvT);
  transpose_w_kernel<<<(kE * kE + 255) / 256, 256, 0, stream>>>(Wo, woT);

  qkv_rope_kernel<<<dim3((kB * kT) / 16, kH, 3), 128, 0, stream>>>(
      xb, wqT, wkT, wvT, q_r, k_r, v_r, sp);

  attn_part_kernel<<<dim3(kH, kB, kParts), 128, 0, stream>>>(
      q_r, k_r, v_r, ck, cv, pacc, pml, sp);

  attn_merge_kernel<<<dim3(kH, kB), 256, 0, stream>>>(pacc, pml, ctx);

  out_gemm_kernel<<<dim3((kB * kT) / 16, kE / 64), 128, 0, stream>>>(ctx, woT, out);
}
